// GRUCell_58205396795594
// MI455X (gfx1250) — matrix operand order repaired and checked by reading, not compiled, not run on hardware
//
#include <hip/hip_runtime.h>
#include <hip/hip_bf16.h>

// ---------------------------------------------------------------------------
// GRU on gfx1250 (MI455X): bf16 WMMA (v_wmma_f32_16x16x32_bf16) everywhere.
//  K1: pack 6 weight matrices fp32 -> bf16 row-major ([n][k]) in workspace.
//  K2: batched input projections xz/xr/xc = x*W^T + b  (M=131072,N=256,K=256)
//  K3: persistent recurrent scan; 1 workgroup per 16 batch rows; Wz_h/Wr_h in
//      LDS (bf16), Wc_h fragments register-resident; h carried in registers
//      (own tile) + LDS bf16 copy for cross-wave WMMA A-fragments.
// ---------------------------------------------------------------------------

#define B_   128
#define S_   1024
#define H_   256
#define ROWP 528   // 512B row + 16B pad (LDS bank spread), 16B aligned

typedef __attribute__((ext_vector_type(16))) __bf16 v16bf;
typedef __attribute__((ext_vector_type(8)))  float  v8f;

union FragU { uint4 u[2]; v16bf v; };

// 32 contiguous bytes (B-fragment: one lane's 16 K-contiguous bf16)
__device__ __forceinline__ v16bf ld_frag_contig(const void* p) {
  FragU f;
  f.u[0] = *(const uint4*)p;
  f.u[1] = *(const uint4*)((const char*)p + 16);
  return f.v;
}
// A-fragment: 32 contiguous bytes (K 16h..16h+15), matching B contiguous order
__device__ __forceinline__ v16bf ld_frag_split(const void* p) {
  FragU f;
  f.u[0] = *(const uint4*)p;
  f.u[1] = *(const uint4*)((const char*)p + 16);
  return f.v;
}

__device__ __forceinline__ v8f wmma_bf16(v16bf a, v16bf b, v8f c) {
  // (neg_a, A, neg_b, B, c_mod, C, reuse_a, reuse_b)
  return __builtin_amdgcn_wmma_f32_16x16x32_bf16(false, a, false, b, (short)0, c,
                                                 false, false);
}

__device__ __forceinline__ float sigmoid_f(float x) {
  return 1.0f / (1.0f + __expf(-x));
}
__device__ __forceinline__ float tanh_f(float x) {
  float e = __expf(-2.0f * fabsf(x));
  float t = (1.0f - e) / (1.0f + e);
  return copysignf(t, x);
}

// ---------------------------------------------------------------------------
// K1: pack weights fp32 -> bf16. Order in ws: Wz_h, Wr_h, Wc_h, Wz_x, Wr_x, Wc_x
// ---------------------------------------------------------------------------
__global__ void pack_weights(const float* __restrict__ wzh, const float* __restrict__ wrh,
                             const float* __restrict__ wch, const float* __restrict__ wzx,
                             const float* __restrict__ wrx, const float* __restrict__ wcx,
                             __bf16* __restrict__ dst) {
  int idx = blockIdx.x * 256 + threadIdx.x;      // 0 .. 6*65536-1
  int m = idx >> 16, off = idx & 65535;
  const float* src = (m == 0) ? wzh : (m == 1) ? wrh : (m == 2) ? wch
                    : (m == 3) ? wzx : (m == 4) ? wrx : wcx;
  dst[idx] = (__bf16)src[off];
}

// ---------------------------------------------------------------------------
// K2: input projections. grid = (2048 Mblocks of 64 rows, 3 gates), 256 thr.
// Output layout xproj[b][s][g][n] bf16 -> byte addr (b*S+s)*1536 + g*512 + 2n
// ---------------------------------------------------------------------------
__global__ void __launch_bounds__(256)
proj_kernel(const float* __restrict__ x, const __bf16* __restrict__ wx_all,
            const float* __restrict__ bz, const float* __restrict__ br,
            const float* __restrict__ bc, __bf16* __restrict__ xproj) {
  extern __shared__ char smem[];
  char* Wlds = smem;                 // 256 * 528 = 135168
  char* xlds = smem + 135168;        //  64 * 528 =  33792
  char* xst  = smem + 168960;        //  64 * 512 =  32768
  const int g    = blockIdx.y;
  const int tid  = threadIdx.x;
  const long Mbase = (long)blockIdx.x * 64;

  // stage gate weight matrix into padded LDS
  const char* wsrc = (const char*)(wx_all + (long)g * 65536);
  for (int q = 0; q < 32; ++q) {
    int c = tid + q * 256;                      // 8192 x 16B
    int row = c >> 5, off = (c & 31) << 4;
    *(uint4*)(Wlds + row * ROWP + off) = *(const uint4*)(wsrc + row * 512 + off);
  }
  // stage x block (fp32 -> bf16)
  for (int q = 0; q < 16; ++q) {
    int c = tid + q * 256;                      // 4096 x float4
    int row = c >> 6, o4 = c & 63;
    float4 v = *(const float4*)(x + (Mbase + row) * 256 + o4 * 4);
    __bf16 h4[4] = {(__bf16)v.x, (__bf16)v.y, (__bf16)v.z, (__bf16)v.w};
    *(uint2*)(xlds + row * ROWP + o4 * 8) = *(uint2*)h4;
  }
  __syncthreads();

  const int wave = tid >> 5, lane = tid & 31;
  const int half = lane >> 4, l16 = lane & 15;
  const float* bias = (g == 0) ? bz : (g == 1) ? br : bc;

  // register-resident B fragments for this wave's two N-tiles
  v16bf wb[2][8];
  float bv[2];
#pragma unroll
  for (int p = 0; p < 2; ++p) {
    int n0 = (2 * wave + p) * 16;
    bv[p] = bias[n0 + l16];
#pragma unroll
    for (int kb = 0; kb < 8; ++kb)
      wb[p][kb] = ld_frag_contig(Wlds + (n0 + l16) * ROWP + kb * 64 + half * 32);
  }

  const char* abase = xlds + l16 * ROWP + half * 32;
  for (int mt = 0; mt < 4; ++mt) {
    v8f acc[2] = {{}, {}};
    v16bf a_cur = ld_frag_split(abase + mt * 16 * ROWP);
#pragma unroll
    for (int kb = 0; kb < 8; ++kb) {
      int kn = (kb < 7) ? kb + 1 : kb;
      v16bf a_nxt = ld_frag_split(abase + mt * 16 * ROWP + kn * 64);
      acc[0] = wmma_bf16(a_cur, wb[0][kb], acc[0]);
      acc[1] = wmma_bf16(a_cur, wb[1][kb], acc[1]);
      a_cur = a_nxt;
    }
#pragma unroll
    for (int p = 0; p < 2; ++p) {
      int n = (2 * wave + p) * 16 + l16;
#pragma unroll
      for (int j = 0; j < 8; ++j) {
        int mrow = mt * 16 + j + 8 * half;
        *(__bf16*)(xst + mrow * 512 + n * 2) = (__bf16)(acc[p][j] + bv[p]);
      }
    }
  }
  __syncthreads();

  // coalesced flush: 64 rows x 512B into strided (1536B) global layout
  for (int q = 0; q < 8; ++q) {
    int c = tid + q * 256;                      // 2048 x 16B
    int row = c >> 5, off = (c & 31) << 4;
    long M = Mbase + row;
    *(uint4*)((char*)xproj + M * 1536 + g * 512 + off) =
        *(const uint4*)(xst + row * 512 + off);
  }
}

// ---------------------------------------------------------------------------
// K3: persistent recurrent scan. grid = 8 (batch tiles of 16), 256 threads.
// ---------------------------------------------------------------------------
__global__ void __launch_bounds__(256)
gru_rec(const __bf16* __restrict__ xproj, const __bf16* __restrict__ whz,
        const __bf16* __restrict__ whr, const __bf16* __restrict__ whc,
        const float* __restrict__ h0, float* __restrict__ out) {
  extern __shared__ char smem[];
  char* Wz = smem;                 // 135168
  char* Wr = smem + 135168;        // -> 270336
  char* hb = smem + 270336;        // 16*528 -> 278784   (h, bf16)
  char* rb = smem + 278784;        // 16*528 -> 287232   (r*h, bf16)
  char* xb = smem + 287232;        // 16*1536 -> 311808  (xz|xr|xc tile, bf16)

  const int tid = threadIdx.x, wave = tid >> 5, lane = tid & 31;
  const int half = lane >> 4, l16 = lane & 15;
  const int b0 = blockIdx.x * 16;

  // stage Wz_h / Wr_h (bf16) into padded LDS
  for (int q = 0; q < 32; ++q) {
    int c = tid + q * 256;
    int row = c >> 5, off = (c & 31) << 4;
    *(uint4*)(Wz + row * ROWP + off) = *(const uint4*)((const char*)whz + row * 512 + off);
    *(uint4*)(Wr + row * ROWP + off) = *(const uint4*)((const char*)whr + row * 512 + off);
  }
  // Wc_h fragments -> registers (persistent across all 1024 steps)
  v16bf wc[2][8];
#pragma unroll
  for (int p = 0; p < 2; ++p) {
    int n0 = (2 * wave + p) * 16;
#pragma unroll
    for (int kb = 0; kb < 8; ++kb)
      wc[p][kb] = ld_frag_contig((const char*)whc + (n0 + l16) * 512 + kb * 64 + half * 32);
  }
  // init h tile in LDS from h0
  for (int q = 0; q < 16; ++q) {
    int c = tid + q * 256;                      // 4096 elements
    int row = c >> 8, n = c & 255;
    *(__bf16*)(hb + row * ROWP + n * 2) = (__bf16)h0[(b0 + row) * 256 + n];
  }
  // carry h for this wave's own (m,n) tile in fp32 registers
  float hreg[2][8];
#pragma unroll
  for (int p = 0; p < 2; ++p) {
    int n = (2 * wave + p) * 16 + l16;
#pragma unroll
    for (int j = 0; j < 8; ++j)
      hreg[p][j] = h0[(b0 + j + 8 * half) * 256 + n];
  }
  __syncthreads();

  const char* habase = hb + l16 * ROWP + half * 32;   // A-frag base (h)
  const char* rabase = rb + l16 * ROWP + half * 32;   // A-frag base (r*h)
  const char* wzbase0 = Wz + ((2 * wave + 0) * 16 + l16) * ROWP + half * 32;
  const char* wzbase1 = Wz + ((2 * wave + 1) * 16 + l16) * ROWP + half * 32;
  const char* wrbase0 = Wr + ((2 * wave + 0) * 16 + l16) * ROWP + half * 32;
  const char* wrbase1 = Wr + ((2 * wave + 1) * 16 + l16) * ROWP + half * 32;

  for (int t = 0; t < S_; ++t) {
    // stage this step's precomputed projections: 16 rows x 1536B
    for (int q = 0; q < 6; ++q) {
      int c = tid + q * 256;                    // 1536 x 16B
      int row = c / 96, off = (c % 96) * 16;
      const char* gsrc =
          (const char*)xproj + ((long)(b0 + row) * S_ + t) * 1536 + off;
      *(uint4*)(xb + row * 1536 + off) = *(const uint4*)gsrc;
      __builtin_prefetch(gsrc + 1536, 0, 1);    // next timestep
    }
    __syncthreads();

    // ---- phase A: z and r gates -----------------------------------------
    v8f az[2] = {{}, {}}, ar[2] = {{}, {}};
    v16bf a_cur = ld_frag_split(habase);
#pragma unroll
    for (int kb = 0; kb < 8; ++kb) {
      int kn = (kb < 7) ? kb + 1 : kb;
      // issue all loads for this (and next A) before the WMMAs
      v16bf bz0 = ld_frag_contig(wzbase0 + kb * 64);
      v16bf bz1 = ld_frag_contig(wzbase1 + kb * 64);
      v16bf br0 = ld_frag_contig(wrbase0 + kb * 64);
      v16bf br1 = ld_frag_contig(wrbase1 + kb * 64);
      v16bf a_nxt = ld_frag_split(habase + kn * 64);
      az[0] = wmma_bf16(a_cur, bz0, az[0]);
      az[1] = wmma_bf16(a_cur, bz1, az[1]);
      ar[0] = wmma_bf16(a_cur, br0, ar[0]);
      ar[1] = wmma_bf16(a_cur, br1, ar[1]);
      a_cur = a_nxt;
    }
    float zv[2][8];
#pragma unroll
    for (int p = 0; p < 2; ++p) {
      int n = (2 * wave + p) * 16 + l16;
#pragma unroll
      for (int j = 0; j < 8; ++j) {
        int m = j + 8 * half;
        float xz = (float)*(const __bf16*)(xb + m * 1536 + 0 * 512 + n * 2);
        float xr = (float)*(const __bf16*)(xb + m * 1536 + 1 * 512 + n * 2);
        float z = sigmoid_f(az[p][j] + xz);
        float r = sigmoid_f(ar[p][j] + xr);
        *(__bf16*)(rb + m * ROWP + n * 2) = (__bf16)(r * hreg[p][j]);
        zv[p][j] = z;
      }
    }
    __syncthreads();

    // ---- phase B: candidate + state update ------------------------------
    v8f ac[2] = {{}, {}};
    v16bf c_cur = ld_frag_split(rabase);
#pragma unroll
    for (int kb = 0; kb < 8; ++kb) {
      int kn = (kb < 7) ? kb + 1 : kb;
      v16bf c_nxt = ld_frag_split(rabase + kn * 64);
      ac[0] = wmma_bf16(c_cur, wc[0][kb], ac[0]);
      ac[1] = wmma_bf16(c_cur, wc[1][kb], ac[1]);
      c_cur = c_nxt;
    }
#pragma unroll
    for (int p = 0; p < 2; ++p) {
      int n = (2 * wave + p) * 16 + l16;
#pragma unroll
      for (int j = 0; j < 8; ++j) {
        int m = j + 8 * half;
        float xc = (float)*(const __bf16*)(xb + m * 1536 + 2 * 512 + n * 2);
        float ht = tanh_f(ac[p][j] + xc);
        float hv = hreg[p][j];
        float hn = hv + zv[p][j] * (ht - hv);
        hreg[p][j] = hn;
        out[((long)(b0 + m) * S_ + t) * H_ + n] = hn;
        *(__bf16*)(hb + m * ROWP + n * 2) = (__bf16)hn;
      }
    }
    __syncthreads();
  }
}

// ---------------------------------------------------------------------------
extern "C" void kernel_launch(void* const* d_in, const int* in_sizes, int n_in,
                              void* d_out, int out_size, void* d_ws, size_t ws_size,
                              hipStream_t stream) {
  const float* x   = (const float*)d_in[0];
  const float* h0  = (const float*)d_in[1];
  const float* Wzx = (const float*)d_in[2];
  const float* bz  = (const float*)d_in[3];
  const float* Wzh = (const float*)d_in[4];
  const float* Wrx = (const float*)d_in[5];
  const float* br  = (const float*)d_in[6];
  const float* Wrh = (const float*)d_in[7];
  const float* Wcx = (const float*)d_in[8];
  const float* bc  = (const float*)d_in[9];
  const float* Wch = (const float*)d_in[10];
  float* out = (float*)d_out;

  char*   ws    = (char*)d_ws;
  __bf16* wpack = (__bf16*)ws;                    // 6 * 65536 bf16 = 768 KB
  __bf16* xproj = (__bf16*)(ws + 786432);         // 131072*3*256 bf16 ~ 192 MB

  // K1: pack weights (order: Wz_h, Wr_h, Wc_h, Wz_x, Wr_x, Wc_x)
  pack_weights<<<1536, 256, 0, stream>>>(Wzh, Wrh, Wch, Wzx, Wrx, Wcx, wpack);

  // K2: input projections for all (b,s): grid (2048 Mblocks, 3 gates)
  dim3 g1(2048, 3);
  proj_kernel<<<g1, 256, 201728, stream>>>(x, wpack + 3 * 65536, bz, br, bc, xproj);

  // K3: persistent recurrent scan: 8 batch tiles of 16 rows
  gru_rec<<<8, 256, 311808, stream>>>(xproj, wpack, wpack + 65536,
                                      wpack + 2 * 65536, h0, out);
}
